// DecoderLayer_47175920779446
// MI455X (gfx1250) — matrix operand order repaired and checked by reading, not compiled, not run on hardware
//
#include <hip/hip_runtime.h>

typedef _Float16 f16;
typedef _Float16 v4h  __attribute__((ext_vector_type(4)));
typedef _Float16 v8h  __attribute__((ext_vector_type(8)));
typedef _Float16 v16h __attribute__((ext_vector_type(16)));
typedef float    v8f  __attribute__((ext_vector_type(8)));

#define DEVFN static __device__ __forceinline__

constexpr int BB = 2, T = 2048, H = 1024, NH = 16, HD = 64, FF = 4096;
constexpr int BT = BB * T;
constexpr float EPS = 1e-6f;

// xor-swizzle across 32 lanes (group-of-32 mode: offset[15]=0, and=0x1f, xor=mask)
#define SWZ_XOR_F(v, mask) \
  __int_as_float(__builtin_amdgcn_ds_swizzle(__float_as_int(v), (0x1f | ((mask) << 10))))

DEVFN v8f vzero() { v8f z = {0.f,0.f,0.f,0.f,0.f,0.f,0.f,0.f}; return z; }

DEVFN v16h ld2(const f16* p0, const f16* p1) {
  v16h r;
  *(v8h*)&r       = *(const v8h*)p0;
  *((v8h*)&r + 1) = *(const v8h*)p1;
  return r;
}

DEVFN v8f wmma_f16(v16h a, v16h b, v8f c) {
  return __builtin_amdgcn_wmma_f32_16x16x32_f16(false, a, false, b, (short)0, c,
                                                false, false);
}

// async global -> LDS copy of 16 bytes per lane (ASYNCcnt-tracked DMA).
// LDS dest address = low 32 bits of the generic shared-memory pointer
// (flat LDS aperture: addr[31:0] is the wave-relative LDS byte offset).
DEVFN void async_cp16(const f16* g, f16* l) {
  unsigned int lds = (unsigned int)(unsigned long long)l;
  unsigned long long ga = (unsigned long long)g;
  asm volatile("global_load_async_to_lds_b128 %0, %1, off"
               :: "v"(lds), "v"(ga) : "memory");
}
DEVFN void wait_async(void) { asm volatile("s_wait_asynccnt 0x0" ::: "memory"); }
DEVFN void wait_async4(void) { asm volatile("s_wait_asynccnt 0x4" ::: "memory"); }

// ---------------------------------------------------------------- convert x
__global__ void k_cvt(const float* __restrict__ in, f16* __restrict__ out, int n) {
  int i = (blockIdx.x * blockDim.x + threadIdx.x) * 4;
  if (i + 3 < n) {
    float4 v = *(const float4*)(in + i);
    v4h o = {(f16)v.x, (f16)v.y, (f16)v.z, (f16)v.w};
    *(v4h*)(out + i) = o;
  }
}

// ----------------------------------------------- transpose+convert: [K,N]f32 -> [N,K]f16
__global__ void k_tcvt(const float* __restrict__ in, f16* __restrict__ out,
                       int K, int N) {
  __shared__ float tile[32][33];
  int n0 = blockIdx.x * 32, k0 = blockIdx.y * 32;
  int tx = threadIdx.x, ty = threadIdx.y;           // (32, 8)
#pragma unroll
  for (int i = 0; i < 4; ++i)
    tile[ty + i * 8][tx] = in[(size_t)(k0 + ty + i * 8) * N + n0 + tx];
  __syncthreads();
#pragma unroll
  for (int i = 0; i < 4; ++i)
    out[(size_t)(n0 + ty + i * 8) * K + k0 + tx] = (f16)tile[tx][ty + i * 8];
}

// ---------------------------------------------------------------- WMMA GEMM
// C[M,N] = A[M,K](f16) x B[K,N], with B supplied transposed Bt[N,K](f16).
// Double-buffered LDS staging via GLOBAL_LOAD_ASYNC_TO_LDS_B128.
// mul-bias epilogue: C *= bias[n].
// MODE 0: scatter into Q[T,HD]/K[T,HD]/Vt[HD,T] per (b,h)  (QKV projection)
// MODE 1: relu -> f16 out[M,N]
// MODE 2: f32 out[M,N]
template <int MODE>
__global__ void k_gemm(const f16* __restrict__ A, const f16* __restrict__ Bt,
                       const float* __restrict__ bias,
                       float* __restrict__ outF32, f16* __restrict__ outF16,
                       f16* __restrict__ Qh, f16* __restrict__ Kh,
                       f16* __restrict__ Vt, int M, int N, int K) {
  __shared__ f16 As[2][128][40];   // +8 halfs pad to spread LDS banks
  __shared__ f16 Bs[2][128][40];

  const int tid = threadIdx.x;                 // 256 threads = 8 waves
  const int wid = tid >> 5, lane = tid & 31;
  const int lr = lane & 15, hs = lane >> 4;
  const int m0 = blockIdx.y * 128, n0 = blockIdx.x * 128;
  const int wm = (wid & 3) * 32;               // wave tile: 32(M) x 64(N)
  const int wn = (wid >> 2) * 64;

  v8f acc[2][4];
#pragma unroll
  for (int i = 0; i < 2; ++i)
#pragma unroll
    for (int j = 0; j < 4; ++j) acc[i][j] = vzero();

  // stage one 128x32 A-tile and B-tile into LDS buffer `buf` (4 async ops/wave)
  auto stage = [&](int buf, int kk) {
#pragma unroll
    for (int c = 0; c < 2; ++c) {
      int cid = tid + c * 256;                 // 512 16B chunks per tile
      int row = cid >> 2, kc = (cid & 3) * 8;
      async_cp16(A  + (size_t)(m0 + row) * K + kk + kc, &As[buf][row][kc]);
      async_cp16(Bt + (size_t)(n0 + row) * K + kk + kc, &Bs[buf][row][kc]);
    }
  };

  stage(0, 0);

  for (int k0 = 0; k0 < K; k0 += 32) {
    const int buf = (k0 >> 5) & 1;
    const bool more = (k0 + 32 < K);
    if (more) stage(buf ^ 1, k0 + 32);         // prefetch next tile (async)
    if (k0 + 64 < K) {                         // warm L2 two tiles ahead
      int row = tid >> 2, kc = (tid & 3) * 8;
      __builtin_prefetch(A  + (size_t)(m0 + row) * K + k0 + 64 + kc, 0, 1);
      __builtin_prefetch(Bt + (size_t)(n0 + row) * K + k0 + 64 + kc, 0, 1);
    }
    if (more) wait_async4();                   // current tile's 4 ops retired
    else      wait_async();
    __syncthreads();                           // all waves' data visible

    v16h af[2], bf[4];
#pragma unroll
    for (int i = 0; i < 2; ++i) {              // A frag: halves {hs*8..}+{16+hs*8..}
      int r = wm + i * 16 + lr;
      af[i] = ld2(&As[buf][r][hs * 8], &As[buf][r][16 + hs * 8]);
    }
#pragma unroll
    for (int j = 0; j < 4; ++j) {              // B frag: halves {hs*8..}+{16+hs*8..}
      int r = wn + j * 16 + lr;
      bf[j] = ld2(&Bs[buf][r][hs * 8], &Bs[buf][r][16 + hs * 8]);
    }
#pragma unroll
    for (int i = 0; i < 2; ++i)
#pragma unroll
      for (int j = 0; j < 4; ++j) acc[i][j] = wmma_f16(af[i], bf[j], acc[i][j]);
    __syncthreads();                           // done reading buf before overwrite
  }

  // epilogue (C layout: VGPR r -> row r+8*hs, col = lane%16)
#pragma unroll
  for (int i = 0; i < 2; ++i) {
#pragma unroll
    for (int j = 0; j < 4; ++j) {
      int ncol = n0 + wn + j * 16 + lr;
      float bv = bias[ncol];
      int mbase = m0 + wm + i * 16 + hs * 8;
      int d = 0, which = 0, hh = 0;
      if (MODE == 0) {
        d = ncol / 48;                 // channel c = d*48 + which*16 + h
        int rem = ncol % 48;
        which = rem >> 4;
        hh = rem & 15;
      }
#pragma unroll
      for (int r = 0; r < 8; ++r) {
        float v = acc[i][j][r] * bv;
        int mrow = mbase + r;
        if (MODE == 0) {
          int b = mrow >> 11, t = mrow & (T - 1);
          if (which == 2)
            Vt[((size_t)((b * NH + hh) * HD + d)) * T + t] = (f16)v;
          else {
            size_t idx = ((size_t)((b * NH + hh) * T + t)) * HD + d;
            (which == 0 ? Qh : Kh)[idx] = (f16)v;
          }
        } else if (MODE == 1) {
          outF16[(size_t)mrow * N + ncol] = (f16)fmaxf(v, 0.f);
        } else {
          outF32[(size_t)mrow * N + ncol] = v;
        }
      }
    }
  }
}

// ------------------------------------------------------- flash attention
// Q,K: [B*NH, T, HD] f16 ; Vt: [B*NH, HD, T] f16 ; out: [B,T,H] f32
// grid (B*NH, T/64), block 128 (4 waves x 16 query rows)
__global__ void k_attn(const f16* __restrict__ Qh, const f16* __restrict__ Kh,
                       const f16* __restrict__ Vt, float* __restrict__ outA) {
  __shared__ f16 Pl[4][16][40];
  const int wid = threadIdx.x >> 5, lane = threadIdx.x & 31;
  const int lr = lane & 15, hs = lane >> 4;
  const int bh = blockIdx.x;
  const int t0 = blockIdx.y * 64 + wid * 16;
  const f16* Qb = Qh + (size_t)bh * T * HD;
  const f16* Kb = Kh + (size_t)bh * T * HD;
  const f16* Vb = Vt + (size_t)bh * HD * T;

  v16h qa[2];
#pragma unroll
  for (int c = 0; c < 2; ++c) {
    const f16* p = Qb + (size_t)(t0 + lr) * HD + c * 32;
    qa[c] = ld2(p + hs * 8, p + 16 + hs * 8);
  }

  v8f o[4];
#pragma unroll
  for (int dn = 0; dn < 4; ++dn) o[dn] = vzero();
  float mrow[8], lrow[8];
#pragma unroll
  for (int r = 0; r < 8; ++r) { mrow[r] = -1e30f; lrow[r] = 0.f; }

  for (int j0 = 0; j0 < T; j0 += 32) {
    v8f s0 = vzero(), s1 = vzero();
#pragma unroll
    for (int c = 0; c < 2; ++c) {              // scores over HD=64 (two K=32 steps)
      const f16* kp0 = Kb + (size_t)(j0 + lr) * HD + c * 32 + hs * 8;
      s0 = wmma_f16(qa[c], ld2(kp0, kp0 + 16), s0);
      const f16* kp1 = Kb + (size_t)(j0 + 16 + lr) * HD + c * 32 + hs * 8;
      s1 = wmma_f16(qa[c], ld2(kp1, kp1 + 16), s1);
    }
#pragma unroll
    for (int r = 0; r < 8; ++r) {              // online softmax, row r+8*hs
      float a = s0[r] * 0.03125f;              // / sqrt(H)
      float b = s1[r] * 0.03125f;
      float mx = fmaxf(a, b);
      mx = fmaxf(mx, SWZ_XOR_F(mx, 1));
      mx = fmaxf(mx, SWZ_XOR_F(mx, 2));
      mx = fmaxf(mx, SWZ_XOR_F(mx, 4));
      mx = fmaxf(mx, SWZ_XOR_F(mx, 8));
      float mn = fmaxf(mrow[r], mx);
      float fac = __expf(mrow[r] - mn);
      float p0 = __expf(a - mn), p1 = __expf(b - mn);
      float sm = p0 + p1;
      sm += SWZ_XOR_F(sm, 1);
      sm += SWZ_XOR_F(sm, 2);
      sm += SWZ_XOR_F(sm, 4);
      sm += SWZ_XOR_F(sm, 8);
      lrow[r] = lrow[r] * fac + sm;
      mrow[r] = mn;
#pragma unroll
      for (int dn = 0; dn < 4; ++dn) o[dn][r] *= fac;
      int prow = r + 8 * hs;
      Pl[wid][prow][lr] = (f16)p0;
      Pl[wid][prow][16 + lr] = (f16)p1;
    }
    // reload P as A-fragment (wave-private LDS region)
    v16h pa = ld2(&Pl[wid][lr][hs * 8], &Pl[wid][lr][16 + hs * 8]);
#pragma unroll
    for (int dn = 0; dn < 4; ++dn) {           // O += P @ V
      const f16* vp = Vb + (size_t)(dn * 16 + lr) * T + j0 + hs * 8;
      o[dn] = wmma_f16(pa, ld2(vp, vp + 16), o[dn]);
    }
  }

  const int b = bh >> 4, hh = bh & 15;
#pragma unroll
  for (int dn = 0; dn < 4; ++dn)
#pragma unroll
    for (int r = 0; r < 8; ++r) {
      int t = t0 + r + 8 * hs;
      outA[((size_t)(b * T + t)) * H + hh * HD + dn * 16 + lr] =
          o[dn][r] / lrow[r];
    }
}

// ------------------------------------------------ residual + layernorm (ddof=1)
__global__ void k_ln(const float* __restrict__ a, const float* __restrict__ r,
                     const float* __restrict__ gamma, const float* __restrict__ beta,
                     float* __restrict__ y32, f16* __restrict__ y16) {
  __shared__ float red1[8], red2[8];
  const int row = blockIdx.x, tid = threadIdx.x;   // 256 threads, H=1024
  const size_t base = (size_t)row * H;
  float x[4], s1 = 0.f, s2 = 0.f;
#pragma unroll
  for (int i = 0; i < 4; ++i) {
    int c = tid + i * 256;
    x[i] = a[base + c] + r[base + c];
    s1 += x[i];
    s2 += x[i] * x[i];
  }
  s1 += SWZ_XOR_F(s1, 1);  s2 += SWZ_XOR_F(s2, 1);
  s1 += SWZ_XOR_F(s1, 2);  s2 += SWZ_XOR_F(s2, 2);
  s1 += SWZ_XOR_F(s1, 4);  s2 += SWZ_XOR_F(s2, 4);
  s1 += SWZ_XOR_F(s1, 8);  s2 += SWZ_XOR_F(s2, 8);
  s1 += SWZ_XOR_F(s1, 16); s2 += SWZ_XOR_F(s2, 16);
  int wid = tid >> 5, lane = tid & 31;
  if (lane == 0) { red1[wid] = s1; red2[wid] = s2; }
  __syncthreads();
  if (tid == 0) {
    float a1 = 0.f, a2 = 0.f;
    for (int w = 0; w < 8; ++w) { a1 += red1[w]; a2 += red2[w]; }
    red1[0] = a1; red2[0] = a2;
  }
  __syncthreads();
  float S1 = red1[0], S2 = red2[0];
  float mean = S1 * (1.f / H);
  float var = (S2 - S1 * mean) / (float)(H - 1);     // unbiased (ddof=1)
  float stdv = sqrtf(fmaxf(var, 0.f));
  float inv = 1.f / (stdv + EPS);
#pragma unroll
  for (int i = 0; i < 4; ++i) {
    int c = tid + i * 256;
    float yv = gamma[c] * (x[i] - mean) * inv + beta[c];
    if (y32) y32[base + c] = yv;
    if (y16) y16[base + c] = (f16)yv;
  }
}

// ---------------------------------------------------------------- launcher
static inline size_t alignup(size_t v) { return (v + 255) & ~(size_t)255; }

extern "C" void kernel_launch(void* const* d_in, const int* in_sizes, int n_in,
                              void* d_out, int out_size, void* d_ws, size_t ws_size,
                              hipStream_t stream) {
  const float* x     = (const float*)d_in[0];
  const float* Wqkv  = (const float*)d_in[1];
  const float* bqkv  = (const float*)d_in[2];
  const float* W1    = (const float*)d_in[3];
  const float* b1    = (const float*)d_in[4];
  const float* W2    = (const float*)d_in[5];
  const float* b2    = (const float*)d_in[6];
  const float* g1    = (const float*)d_in[7];
  const float* be1   = (const float*)d_in[8];
  const float* g2    = (const float*)d_in[9];
  const float* be2   = (const float*)d_in[10];
  float* out = (float*)d_out;

  char* p = (char*)d_ws;
  auto take = [&](size_t bytes) { char* q = p; p += alignup(bytes); return q; };
  f16*   xh      = (f16*)  take((size_t)BT * H * 2);
  f16*   WqkvT   = (f16*)  take((size_t)3 * H * H * 2);
  f16*   W1T     = (f16*)  take((size_t)FF * H * 2);
  f16*   W2T     = (f16*)  take((size_t)H * FF * 2);
  f16*   Qh      = (f16*)  take((size_t)BB * NH * T * HD * 2);
  f16*   Kh      = (f16*)  take((size_t)BB * NH * T * HD * 2);
  f16*   Vt      = (f16*)  take((size_t)BB * NH * HD * T * 2);
  float* attnout = (float*)take((size_t)BT * H * 4);
  float* ln1f    = (float*)take((size_t)BT * H * 4);
  f16*   ln1h    = (f16*)  take((size_t)BT * H * 2);
  f16*   mid     = (f16*)  take((size_t)BT * FF * 2);
  float* ffnout  = (float*)take((size_t)BT * H * 4);

  // 1. convert x -> f16
  k_cvt<<<dim3(BT * H / 1024), dim3(256), 0, stream>>>(x, xh, BT * H);
  // 2. weights: transpose + convert to [N,K] f16
  k_tcvt<<<dim3(3 * H / 32, H / 32),  dim3(32, 8), 0, stream>>>(Wqkv, WqkvT, H, 3 * H);
  k_tcvt<<<dim3(FF / 32, H / 32),     dim3(32, 8), 0, stream>>>(W1,   W1T,   H, FF);
  k_tcvt<<<dim3(H / 32, FF / 32),     dim3(32, 8), 0, stream>>>(W2,   W2T,   FF, H);
  // 3. QKV projection (mul-bias) + scatter to per-head Q/K/Vt
  k_gemm<0><<<dim3(3 * H / 128, BT / 128), dim3(256), 0, stream>>>(
      xh, WqkvT, bqkv, nullptr, nullptr, Qh, Kh, Vt, BT, 3 * H, H);
  // 4. flash attention
  k_attn<<<dim3(BB * NH, T / 64), dim3(128), 0, stream>>>(Qh, Kh, Vt, attnout);
  // 5. residual + LN1
  k_ln<<<dim3(BT), dim3(256), 0, stream>>>(attnout, x, g1, be1, ln1f, ln1h);
  // 6. FFN1 (mul-bias + relu)
  k_gemm<1><<<dim3(FF / 128, BT / 128), dim3(256), 0, stream>>>(
      ln1h, W1T, b1, nullptr, mid, nullptr, nullptr, nullptr, BT, FF, H);
  // 7. FFN2 (mul-bias)
  k_gemm<2><<<dim3(H / 128, BT / 128), dim3(256), 0, stream>>>(
      mid, W2T, b2, ffnout, nullptr, nullptr, nullptr, nullptr, BT, H, FF);
  // 8. residual + LN2 -> output
  k_ln<<<dim3(BT), dim3(256), 0, stream>>>(ffnout, ln1f, g2, be2, out, nullptr);
}
